// Attention_76802605187583
// MI455X (gfx1250) — matrix operand order repaired and checked by reading, not compiled, not run on hardware
//
#include <hip/hip_runtime.h>

#define HEADS 8
#define BATCH 4
#define SEQ   2048
#define DTOT  4096
#define DOUTC 512
#define MTOT  (BATCH * SEQ)   // 8192

typedef __attribute__((ext_vector_type(16))) __bf16 v16bf;
typedef __attribute__((ext_vector_type(8)))  float  v8f;

static __device__ __forceinline__ unsigned short f2bf(float f) {
  union { float f; unsigned u; } c; c.f = f;
  unsigned u = c.u;
  u += 0x7FFFu + ((u >> 16) & 1u);      // round-to-nearest-even
  return (unsigned short)(u >> 16);
}

// CDNA5 async memory->LDS copy (ASYNCcnt-tracked, no VGPR data movement).
// VDST = VGPR holding LDS byte address, VADDR = 64-bit global address (GV mode).
static __device__ __forceinline__ void async_ld_b128(unsigned lds_addr, const void* gaddr) {
  asm volatile("global_load_async_to_lds_b128 %0, %1, off"
               :: "v"(lds_addr), "v"(gaddr) : "memory");
}
static __device__ __forceinline__ void async_wait0() {
  asm volatile("s_wait_asynccnt 0" ::: "memory");
}
// LDS byte offset of a shared-memory pointer (low 32 bits of the generic address)
template <typename T>
static __device__ __forceinline__ unsigned lds_off(T* p) {
  return (unsigned)(uintptr_t)p;
}

// A fragment: 16x32 bf16, source row-major [row][k] in LDS.
static __device__ __forceinline__ v16bf frag_a(const unsigned short* base, int row0,
                                               int stride, int col0) {
  const int lane = threadIdx.x & 31;
  const int m = lane & 15, half = lane >> 4;
  const unsigned short* rp = base + (row0 + m) * stride + col0;
  union { v16bf v; unsigned u[8]; } r;
#pragma unroll
  for (int g = 0; g < 8; ++g) {
    int k = (g < 4 ? 2 * g : 16 + 2 * (g - 4)) + 8 * half;
    __builtin_memcpy(&r.u[g], rp + k, 4);
  }
  return r.v;
}

// B fragment: 32x16 bf16; source stored as B^T row-major [n][k] in LDS.
static __device__ __forceinline__ v16bf frag_b(const unsigned short* base, int row0,
                                               int stride, int col0) {
  const int lane = threadIdx.x & 31;
  const int n = lane & 15, half = lane >> 4;
  const unsigned short* rp = base + (row0 + n) * stride + col0 + half * 8;
  union { v16bf v; unsigned u[8]; } r;
#pragma unroll
  for (int g = 0; g < 8; ++g) __builtin_memcpy(&r.u[g], rp + (g < 4 ? 2 * g : 16 + 2 * (g - 4)), 4);
  return r.v;
}

#define WMMA_BF16(a, b, c) \
  __builtin_amdgcn_wmma_f32_16x16x32_bf16(false, (a), false, (b), (short)0, (c), false, false)

// ---------------------------------------------------------------- utilities
__global__ void cvt_f32_bf16(const float* __restrict__ src,
                             unsigned short* __restrict__ dst, int n) {
  int i = blockIdx.x * blockDim.x + threadIdx.x;
  if (i < n) dst[i] = f2bf(src[i]);
}

__global__ void zero_u16(unsigned short* __restrict__ p, int n) {
  int i = blockIdx.x * blockDim.x + threadIdx.x;
  if (i < n) p[i] = 0;
}

// ------------------------------------------------- QKV projection GEMM
// C[8192, ncols] = x[:, xoff:xoff+Kdim] (fp32->bf16) @ W^T, scattered into
// Qcat[b,h,n,96] / Kcat[b,h,n,96] / Vcat[bh][chunk][col(64)][key(40 padded)].
__global__ __launch_bounds__(256) void qkv_gemm(
    const float* __restrict__ x, const unsigned short* __restrict__ w,
    unsigned short* __restrict__ qc, unsigned short* __restrict__ kc,
    unsigned short* __restrict__ vc,
    int Kdim, int xoff, int sec_sh, int dh_sh, int qseg, int vseg) {
  __shared__ __align__(16) unsigned short alds[64][40];
  __shared__ __align__(16) unsigned short blds[64][40];
  const int tid = threadIdx.x;
  const int m0 = blockIdx.y * 64;
  const int n0 = blockIdx.x * 64;
  const int wave = tid >> 5, lane = tid & 31;
  const int wm = wave & 3, wn = wave >> 2;
  const int half = lane >> 4, l16 = lane & 15;
  const unsigned bbase = lds_off(&blds[0][0]);

  v8f c0 = {}, c1 = {};
  for (int kb = 0; kb < Kdim; kb += 32) {
    // async copy B tile: 64 rows x 64 bytes into padded [64][40] LDS image
    {
      int row = tid >> 2, colB = (tid & 3) * 16;
      const char* g = (const char*)w + ((size_t)(n0 + row) * Kdim + kb) * 2 + colB;
      async_ld_b128(bbase + row * 80 + colB, g);
    }
    if (kb + 32 < Kdim)
      __builtin_prefetch(x + (size_t)(m0 + (tid >> 3)) * DTOT + xoff + kb + 32, 0, 1);
#pragma unroll
    for (int j = 0; j < 8; ++j) {          // A tile with fp32->bf16 conversion
      int e = tid + j * 256;
      int row = e >> 5, col = e & 31;
      alds[row][col] = f2bf(x[(size_t)(m0 + row) * DTOT + xoff + kb + col]);
    }
    async_wait0();
    __syncthreads();
    v16bf a  = frag_a(&alds[0][0], wm * 16, 40, 0);
    v16bf b0 = frag_b(&blds[0][0], wn * 32, 40, 0);
    v16bf b1 = frag_b(&blds[0][0], wn * 32 + 16, 40, 0);
    c0 = WMMA_BF16(a, b0, c0);
    c1 = WMMA_BF16(a, b1, c1);
    __syncthreads();
  }
#pragma unroll
  for (int t = 0; t < 2; ++t) {
    v8f cc = t ? c1 : c0;
#pragma unroll
    for (int r = 0; r < 8; ++r) {
      int gm = m0 + wm * 16 + r + 8 * half;            // global row in [0,8192)
      int gc = n0 + wn * 32 + t * 16 + l16;            // global output column
      int sec  = gc >> sec_sh;                         // 0=q 1=k 2=v
      int cw   = gc & ((1 << sec_sh) - 1);
      int head = cw >> dh_sh;
      int dim  = cw & ((1 << dh_sh) - 1);
      int b = gm >> 11, n = gm & (SEQ - 1);
      unsigned short v = f2bf(cc[r]);
      if (sec == 2) {                                  // transposed, chunk-padded V
        size_t vi = ((size_t)(b * HEADS + head) * 64 + (n >> 5)) * 2560 +
                    (size_t)(vseg + dim) * 40 + (n & 31);
        vc[vi] = v;
      } else {
        size_t basei = (size_t)(b * HEADS + head) * SEQ + n;
        if (sec == 0) qc[basei * 96 + qseg + dim] = v;
        else          kc[basei * 96 + qseg + dim] = v;
      }
    }
  }
}

// ------------------------------------------------- flash-style 3-branch attention
__global__ __launch_bounds__(256) void attn_kernel(
    const unsigned short* __restrict__ qc, const unsigned short* __restrict__ kc,
    const unsigned short* __restrict__ vc, const int* __restrict__ mask,
    unsigned short* __restrict__ ob) {
  __shared__ __align__(16) unsigned short qlds[128][104]; // query tile, padded stride
  __shared__ __align__(16) unsigned short klds[32][104];  // key chunk
  __shared__ __align__(16) unsigned short vlds[64][40];   // V chunk [col][key(+pad)]
  __shared__ __align__(16) int            mlds[128][32];  // mask chunk
  __shared__ __align__(16) unsigned short plds[8][16][32];// per-wave P scratch

  const int tid  = threadIdx.x;
  const int bh   = blockIdx.y;
  const int b    = bh >> 3, h = bh & 7;
  const int q0   = blockIdx.x * 128;
  const int wave = tid >> 5, lane = tid & 31;
  const int half = lane >> 4, l16 = lane & 15;
  const size_t headbase = (size_t)(b * HEADS + h) * SEQ;
  const unsigned qbase = lds_off(&qlds[0][0]);
  const unsigned kbase = lds_off(&klds[0][0]);
  const unsigned vbase = lds_off(&vlds[0][0]);
  const unsigned mbase = lds_off(&mlds[0][0]);

  // async load Q tile: 128 rows x 192B (12 x 16B chunks per row)
#pragma unroll
  for (int j = 0; j < 6; ++j) {
    int idx = tid + j * 256;                 // 0..1535
    int row = idx / 12, colB = (idx % 12) * 16;
    const char* g = (const char*)qc + (headbase + q0 + row) * 192 + colB;
    async_ld_b128(qbase + row * 208 + colB, g);
  }

  float mr[3][8], lr[3][8];
  v8f oacc[3][4];
  const v8f vzero = {};
#pragma unroll
  for (int br = 0; br < 3; ++br) {
#pragma unroll
    for (int r = 0; r < 8; ++r) { mr[br][r] = -3.0e38f; lr[br][r] = 0.0f; }
#pragma unroll
    for (int t = 0; t < 4; ++t) oacc[br][t] = vzero;
  }
  const float scales[3] = {0.02209708691207961f, 0.03125f, 0.03125f}; // DA^-.5, DP^-.5, DK^-.5
  const int qb = wave * 16;

  for (int kb = 0; kb < SEQ; kb += 32) {
    __syncthreads();   // previous chunk fully consumed before overwrite
    // K chunk: 32 rows x 192B
#pragma unroll
    for (int j = 0; j < 2; ++j) {
      int idx = tid + j * 256;
      if (idx < 384) {
        int row = idx / 12, colB = (idx % 12) * 16;
        const char* g = (const char*)kc + (headbase + kb + row) * 192 + colB;
        async_ld_b128(kbase + row * 208 + colB, g);
      }
    }
    // V chunk: contiguous padded image, 5120B
    {
      const char* vchunk =
          (const char*)vc + (((size_t)(b * HEADS + h) * 64 + (kb >> 5)) * 2560) * 2;
#pragma unroll
      for (int j = 0; j < 2; ++j) {
        int idx = tid + j * 256;
        if (idx < 320) async_ld_b128(vbase + idx * 16, vchunk + idx * 16);
      }
    }
    // mask chunk: 128 rows x 128B
#pragma unroll
    for (int j = 0; j < 4; ++j) {
      int idx = tid + j * 256;               // 0..1023
      int row = idx >> 3, colB = (idx & 7) * 16;
      const char* g = (const char*)mask +
                      ((size_t)b * SEQ * SEQ + (size_t)(q0 + row) * SEQ + kb) * 4 + colB;
      async_ld_b128(mbase + row * 128 + colB, g);
    }
    async_wait0();
    __syncthreads();

#pragma unroll
    for (int br = 0; br < 3; ++br) {
      const int seg = br * 32;
      const float scale = scales[br];
      v16bf a   = frag_a(&qlds[0][0], qb, 104, seg);
      v16bf bk0 = frag_b(&klds[0][0], 0, 104, seg);
      v16bf bk1 = frag_b(&klds[0][0], 16, 104, seg);
      v8f s0 = vzero, s1 = vzero;
      s0 = WMMA_BF16(a, bk0, s0);
      s1 = WMMA_BF16(a, bk1, s1);
      float cr[8];
#pragma unroll
      for (int r = 0; r < 8; ++r) {
        int qrow = qb + r + 8 * half;
        float t0 = (mlds[qrow][l16]      > 0) ? s0[r] * scale : -1.0e10f;
        float t1 = (mlds[qrow][16 + l16] > 0) ? s1[r] * scale : -1.0e10f;
        float mx = fmaxf(t0, t1);
        mx = fmaxf(mx, __shfl_xor(mx, 8, 32));
        mx = fmaxf(mx, __shfl_xor(mx, 4, 32));
        mx = fmaxf(mx, __shfl_xor(mx, 2, 32));
        mx = fmaxf(mx, __shfl_xor(mx, 1, 32));
        float mnew = fmaxf(mr[br][r], mx);
        cr[r] = __expf(mr[br][r] - mnew);
        mr[br][r] = mnew;
        float e0 = __expf(t0 - mnew);
        float e1 = __expf(t1 - mnew);
        plds[wave][r + 8 * half][l16]      = f2bf(e0);
        plds[wave][r + 8 * half][16 + l16] = f2bf(e1);
        float sum = e0 + e1;
        sum += __shfl_xor(sum, 8, 32);
        sum += __shfl_xor(sum, 4, 32);
        sum += __shfl_xor(sum, 2, 32);
        sum += __shfl_xor(sum, 1, 32);
        lr[br][r] = lr[br][r] * cr[r] + sum;
      }
      // same-wave LDS ops are in-order: safe to reload P without a block barrier
      v16bf pa = frag_a(&plds[wave][0][0], 0, 32, 0);
#pragma unroll
      for (int vt = 0; vt < 4; ++vt) {
        v16bf bv = frag_b(&vlds[0][0], vt * 16, 40, 0);
        v8f o = oacc[br][vt];
#pragma unroll
        for (int r = 0; r < 8; ++r) o[r] *= cr[r];
        oacc[br][vt] = WMMA_BF16(pa, bv, o);
      }
    }
  }

#pragma unroll
  for (int vt = 0; vt < 4; ++vt) {
#pragma unroll
    for (int r = 0; r < 8; ++r) {
      int n = q0 + qb + r + 8 * half;
      int col = vt * 16 + l16;
      float val = oacc[0][vt][r] / lr[0][r] + oacc[1][vt][r] / lr[1][r] +
                  oacc[2][vt][r] / lr[2][r];
      val *= (1.0f / 3.0f);
      ob[((size_t)(b * SEQ + n) * HEADS + h) * 64 + col] = f2bf(val);
    }
  }
}

// ------------------------------------------------- output projection + bias
__global__ __launch_bounds__(256) void out_gemm(
    const unsigned short* __restrict__ aab, const unsigned short* __restrict__ w,
    const float* __restrict__ bias, float* __restrict__ out) {
  __shared__ __align__(16) unsigned short alds[64][40];
  __shared__ __align__(16) unsigned short blds[64][40];
  const int tid = threadIdx.x;
  const int m0 = blockIdx.y * 64, n0 = blockIdx.x * 64;
  const int wave = tid >> 5, lane = tid & 31;
  const int wm = wave & 3, wn = wave >> 2;
  const int half = lane >> 4, l16 = lane & 15;
  const unsigned abase = lds_off(&alds[0][0]);
  const unsigned bbase = lds_off(&blds[0][0]);

  v8f c0 = {}, c1 = {};
  for (int kb = 0; kb < DOUTC; kb += 32) {
    {
      int row = tid >> 2, colB = (tid & 3) * 16;
      const char* ga = (const char*)aab + ((size_t)(m0 + row) * DOUTC + kb) * 2 + colB;
      const char* gb = (const char*)w   + ((size_t)(n0 + row) * DOUTC + kb) * 2 + colB;
      async_ld_b128(abase + row * 80 + colB, ga);
      async_ld_b128(bbase + row * 80 + colB, gb);
    }
    async_wait0();
    __syncthreads();
    v16bf a  = frag_a(&alds[0][0], wm * 16, 40, 0);
    v16bf b0 = frag_b(&blds[0][0], wn * 32, 40, 0);
    v16bf b1 = frag_b(&blds[0][0], wn * 32 + 16, 40, 0);
    c0 = WMMA_BF16(a, b0, c0);
    c1 = WMMA_BF16(a, b1, c1);
    __syncthreads();
  }
#pragma unroll
  for (int t = 0; t < 2; ++t) {
    v8f cc = t ? c1 : c0;
#pragma unroll
    for (int r = 0; r < 8; ++r) {
      int gm = m0 + wm * 16 + r + 8 * half;
      int gc = n0 + wn * 32 + t * 16 + l16;
      out[(size_t)gm * DOUTC + gc] = cc[r] + bias[gc];
    }
  }
}

// ------------------------------------------------- host orchestration
extern "C" void kernel_launch(void* const* d_in, const int* in_sizes, int n_in,
                              void* d_out, int out_size, void* d_ws, size_t ws_size,
                              hipStream_t stream) {
  const float* x  = (const float*)d_in[0];
  const float* Wa = (const float*)d_in[1];
  const float* Wp = (const float*)d_in[2];
  const float* Wk = (const float*)d_in[3];
  const float* Wo = (const float*)d_in[4];
  const float* bo = (const float*)d_in[5];
  const int*  msk = (const int*)d_in[6];
  float* out = (float*)d_out;

  char* ws = (char*)d_ws;
  size_t off = 0;
  auto alloc = [&](size_t bytes) -> void* {
    void* p = ws + off;
    off += (bytes + 255) & ~(size_t)255;
    return p;
  };
  unsigned short* wa_b = (unsigned short*)alloc((size_t)768 * 2048 * 2);
  unsigned short* wp_b = (unsigned short*)alloc((size_t)384 * 1024 * 2);
  unsigned short* wk_b = (unsigned short*)alloc((size_t)384 * 1024 * 2);
  unsigned short* wo_b = (unsigned short*)alloc((size_t)512 * 512 * 2);
  unsigned short* qcat = (unsigned short*)alloc((size_t)BATCH * HEADS * SEQ * 96 * 2);
  unsigned short* kcat = (unsigned short*)alloc((size_t)BATCH * HEADS * SEQ * 96 * 2);
  unsigned short* vcat = (unsigned short*)alloc((size_t)BATCH * HEADS * 64 * 2560 * 2);
  unsigned short* obuf = (unsigned short*)alloc((size_t)MTOT * DOUTC * 2);

  auto cvt = [&](const float* s, unsigned short* d, int n) {
    cvt_f32_bf16<<<(n + 255) / 256, 256, 0, stream>>>(s, d, n);
  };
  cvt(Wa, wa_b, 768 * 2048);
  cvt(Wp, wp_b, 384 * 1024);
  cvt(Wk, wk_b, 384 * 1024);
  cvt(Wo, wo_b, 512 * 512);

  const int qn = BATCH * HEADS * SEQ * 96;
  zero_u16<<<(qn + 255) / 256, 256, 0, stream>>>(qcat, qn);
  zero_u16<<<(qn + 255) / 256, 256, 0, stream>>>(kcat, qn);

  // branch a: K=2048, 768 cols (sec width 256 -> shift 8, head dim 32 -> shift 5)
  qkv_gemm<<<dim3(12, MTOT / 64), 256, 0, stream>>>(x, wa_b, qcat, kcat, vcat,
                                                    2048, 0, 8, 5, 0, 0);
  // branch p: K=1024, 384 cols (sec width 128 -> shift 7, head dim 16 -> shift 4)
  qkv_gemm<<<dim3(6, MTOT / 64), 256, 0, stream>>>(x, wp_b, qcat, kcat, vcat,
                                                   1024, 2048, 7, 4, 32, 32);
  // branch k
  qkv_gemm<<<dim3(6, MTOT / 64), 256, 0, stream>>>(x, wk_b, qcat, kcat, vcat,
                                                   1024, 3072, 7, 4, 64, 48);

  attn_kernel<<<dim3(SEQ / 128, BATCH * HEADS), 256, 0, stream>>>(qcat, kcat, vcat,
                                                                  msk, obuf);
  out_gemm<<<dim3(DOUTC / 64, MTOT / 64), 256, 0, stream>>>(obuf, wo_b, bo, out);
}
